// MegaSentenceEncoderLayer_46067819217271
// MI455X (gfx1250) — compile-verified, matrix operand order checked by reading, not run on hardware
//
#include <hip/hip_runtime.h>
#include <hip/hip_bf16.h>

typedef _Float16 v16h __attribute__((ext_vector_type(16)));
typedef _Float16 v8h  __attribute__((ext_vector_type(8)));
typedef float    v8f  __attribute__((ext_vector_type(8)));

#define LSEQ   2048
#define BATCH  16
#define DMODEL 512
#define HDIM   1024
#define ZDIM   128
#define NCHUNK 16
#define CHUNKSZ 128
#define NTOTAL 2176          // Z + H + 2D
#define MROWS  32768         // LSEQ * BATCH
#define EPSV   1e-6f

__device__ __forceinline__ float sigf(float x)  { return 1.f / (1.f + __expf(-x)); }
__device__ __forceinline__ float siluf(float x) { return x * sigf(x); }

__device__ __forceinline__ v8f wmma_f16(v16h a, v16h b, v8f c) {
  return __builtin_amdgcn_wmma_f32_16x16x32_f16(false, a, false, b, (short)0, c, false, false);
}

// Low 32 bits of a flat pointer to LDS == LDS byte offset (AS3->generic cast is
// {shared_aperture_hi, offset}), which is what VDST of the async-to-LDS ops wants.
__device__ __forceinline__ unsigned lds_off_of(const void* p) {
  return (unsigned)(unsigned long long)p;
}

// CDNA5 async DMA: 16B per lane, global -> LDS, tracked by ASYNCcnt.
__device__ __forceinline__ void async_ld16(unsigned lds_off, const void* gaddr) {
  asm volatile("global_load_async_to_lds_b128 %0, %1, off"
               :: "v"(lds_off), "v"(gaddr) : "memory");
}
__device__ __forceinline__ void wait_async0() {
  asm volatile("s_wait_asynccnt 0" ::: "memory");
}

// Load a 16x32 f16 operand fragment from an LDS tile (row-major, row_stride halves).
// Lanes 0-15: row=lane, K={0..7,16..23}; lanes 16-31: row=lane-16, K={8..15,24..31}.
__device__ __forceinline__ v16h load_frag(const _Float16* lds, int row0, int row_stride, int kbase) {
  int lane = threadIdx.x & 31;
  int r = row0 + (lane & 15);
  int kb = kbase + ((lane >> 4) << 3);
  const _Float16* p = lds + (size_t)r * row_stride + kb;
  v8h lo = *(const v8h*)p;
  v8h hi = *(const v8h*)(p + 16);
  return __builtin_shufflevector(lo, hi, 0,1,2,3,4,5,6,7,8,9,10,11,12,13,14,15);
}

// ---------------------------------------------------------------- utilities
__global__ void convert_to_f16(const float* __restrict__ in, _Float16* __restrict__ out, size_t n) {
  for (size_t i = (size_t)blockIdx.x * blockDim.x + threadIdx.x; i < n;
       i += (size_t)gridDim.x * blockDim.x)
    out[i] = (_Float16)in[i];
}

// in: (rows, cols) f32 row-major  ->  out: (cols, rows) f16 row-major (N-major for WMMA B)
__global__ void transpose_to_f16(const float* __restrict__ in, _Float16* __restrict__ out,
                                 int rows, int cols) {
  size_t total = (size_t)rows * cols;
  for (size_t i = (size_t)blockIdx.x * blockDim.x + threadIdx.x; i < total;
       i += (size_t)gridDim.x * blockDim.x) {
    size_t r = i / cols, c = i % cols;
    out[c * (size_t)rows + r] = (_Float16)in[i];
  }
}

// ---------------------------------------------------------------- EMA (scan form of the FFT conv)
__global__ __launch_bounds__(128)
void ema_kernel(const float* __restrict__ x, const float* __restrict__ delta,
                const float* __restrict__ alpha, const float* __restrict__ beta,
                const float* __restrict__ gamma, const float* __restrict__ omega,
                float* __restrict__ fwdbuf, _Float16* __restrict__ mx16) {
  int tid = blockIdx.x * blockDim.x + threadIdx.x;
  if (tid >= BATCH * DMODEL) return;
  int b = tid / DMODEL, d = tid % DMODEL;
  const float invsq = 0.70710678118f; // 1/sqrt(NDIM)

  float q1[2], w1[2], q2[2], w2[2];
#pragma unroll
  for (int n = 0; n < 2; n++) {
    int i1 = d * 2 + n;
    float p1 = sigf(delta[i1]);
    q1[n] = 1.f - p1 * sigf(alpha[i1]);
    w1[n] = p1 * beta[i1] * gamma[i1] * invsq;
    int i2 = (d + DMODEL) * 2 + n;
    float p2 = sigf(delta[i2]);
    q2[n] = 1.f - p2 * sigf(alpha[i2]);
    w2[n] = p2 * beta[i2] * gamma[i2] * invsq;
  }

  const size_t stride = (size_t)BATCH * DMODEL;
  const size_t base = (size_t)b * DMODEL + d;
  float f0 = 0.f, f1 = 0.f;
  for (int t = 0; t < LSEQ; t++) {
    float xv = x[base + (size_t)t * stride];
    f0 = xv + q1[0] * f0;
    f1 = xv + q1[1] * f1;
    fwdbuf[base + (size_t)t * stride] = w1[0] * f0 + w1[1] * f1;
  }
  float g0 = 0.f, g1 = 0.f;
  float om = omega[d];
  for (int t = LSEQ - 1; t >= 0; t--) {
    float xv = x[base + (size_t)t * stride];
    g0 = xv + q2[0] * g0;
    g1 = xv + q2[1] * g1;
    float pre = fwdbuf[base + (size_t)t * stride] + w2[0] * g0 + w2[1] * g1 + xv * om;
    mx16[base + (size_t)t * stride] = (_Float16)siluf(pre);
  }
}

// ---------------------------------------------------------------- WMMA GEMM  C = A(f16) * B^T(f16)
// 128 thr = 4 waves; WG tile 64x64, wave tile 32x32, K step 32.
// Double-buffered async global->LDS staging (ASYNCcnt) overlapped with WMMA.
template <int MODE>
__global__ __launch_bounds__(128)
void gemm_kernel(const _Float16* __restrict__ Ag, const _Float16* __restrict__ Bt,
                 int Msz, int Nsz, int Ksz, const float* __restrict__ bias,
                 _Float16* __restrict__ out16,
                 float* __restrict__ outU, _Float16* __restrict__ outQ,
                 _Float16* __restrict__ outK, _Float16* __restrict__ outR,
                 float* __restrict__ outHX,
                 const float* __restrict__ qkg, const float* __restrict__ qkb,
                 const float* __restrict__ hxin, const float* __restrict__ uin,
                 const float* __restrict__ xres, float* __restrict__ outPN) {
  __shared__ __align__(16) _Float16 As[2][64 * 32];
  __shared__ __align__(16) _Float16 Bs[2][64 * 32];

  const int tid = threadIdx.x;
  const int wave = tid >> 5, lane = tid & 31;
  const int wm = (wave >> 1) * 32, wn = (wave & 1) * 32;
  const int m0 = blockIdx.y * 64, n0 = blockIdx.x * 64;

  v8f acc[2][2] = {};

  const int lr = tid >> 1, lko = (tid & 1) * 16;
  const _Float16* rowA = Ag + (size_t)(m0 + lr) * Ksz + lko;
  const _Float16* rowB = Bt + (size_t)(n0 + lr) * Ksz + lko;

  // prologue: stage tile 0 via async DMA
  async_ld16(lds_off_of(&As[0][lr * 32 + lko]),     rowA);
  async_ld16(lds_off_of(&As[0][lr * 32 + lko + 8]), rowA + 8);
  async_ld16(lds_off_of(&Bs[0][lr * 32 + lko]),     rowB);
  async_ld16(lds_off_of(&Bs[0][lr * 32 + lko + 8]), rowB + 8);
  wait_async0();
  __syncthreads();

  int cur = 0;
  for (int k0 = 0; k0 < Ksz; k0 += 32) {
    if (k0 + 32 < Ksz) { // stage next tile while computing on current
      int nb = cur ^ 1;
      async_ld16(lds_off_of(&As[nb][lr * 32 + lko]),     rowA + k0 + 32);
      async_ld16(lds_off_of(&As[nb][lr * 32 + lko + 8]), rowA + k0 + 40);
      async_ld16(lds_off_of(&Bs[nb][lr * 32 + lko]),     rowB + k0 + 32);
      async_ld16(lds_off_of(&Bs[nb][lr * 32 + lko + 8]), rowB + k0 + 40);
      if (k0 + 64 < Ksz) { // gfx1250 global_prefetch_b8, two tiles ahead
        __builtin_prefetch(rowA + k0 + 64, 0, 1);
        __builtin_prefetch(rowB + k0 + 64, 0, 1);
      }
    }
    v16h a0 = load_frag(As[cur], wm + 0, 32, 0);
    v16h a1 = load_frag(As[cur], wm + 16, 32, 0);
    v16h b0 = load_frag(Bs[cur], wn + 0, 32, 0);
    v16h b1 = load_frag(Bs[cur], wn + 16, 32, 0);
    acc[0][0] = wmma_f16(a0, b0, acc[0][0]);
    acc[0][1] = wmma_f16(a0, b1, acc[0][1]);
    acc[1][0] = wmma_f16(a1, b0, acc[1][0]);
    acc[1][1] = wmma_f16(a1, b1, acc[1][1]);
    wait_async0();
    __syncthreads();
    cur ^= 1;
  }

  const int half = lane >> 4, ln = lane & 15;
#pragma unroll
  for (int mt = 0; mt < 2; mt++)
#pragma unroll
    for (int nt = 0; nt < 2; nt++)
#pragma unroll
      for (int j = 0; j < 8; j++) {
        int m = m0 + wm + mt * 16 + half * 8 + j;
        int n = n0 + wn + nt * 16 + ln;
        float val = acc[mt][nt][j] + bias[n];
        if constexpr (MODE == 0) {
          out16[(size_t)m * Nsz + n] = (_Float16)siluf(val);
        } else if constexpr (MODE == 1) {
          if (n < DMODEL) {
            outU[(size_t)m * DMODEL + n] = sigf(val);
          } else if (n < DMODEL + ZDIM) {
            int zi = n - DMODEL;
            float s = siluf(val);
            outQ[(size_t)m * ZDIM + zi] =
                (_Float16)((s * qkg[zi] + qkb[zi]) * 0.08838834765f); // * Z^-0.5
            outK[(size_t)m * ZDIM + zi] = (_Float16)(s * qkg[ZDIM + zi] + qkb[ZDIM + zi]);
          } else if (n < DMODEL + ZDIM + HDIM) {
            outR[(size_t)m * HDIM + (n - DMODEL - ZDIM)] = (_Float16)siluf(val);
          } else {
            outHX[(size_t)m * DMODEL + (n - DMODEL - ZDIM - HDIM)] = val;
          }
        } else { // MODE == 2
          float t = siluf(hxin[(size_t)m * DMODEL + n] + val);
          float res = xres[(size_t)m * DMODEL + n];
          float u = uin[(size_t)m * DMODEL + n];
          outPN[(size_t)m * DMODEL + n] = res + u * (t - res);
        }
      }
}

// ---------------------------------------------------------------- attention per (b, chunk)
// 256 thr = 8 waves; wave w owns row strip [16w, 16w+16) of the 128x128 score tile.
__global__ __launch_bounds__(256)
void attn_kernel(const _Float16* __restrict__ q16, const _Float16* __restrict__ k16,
                 const _Float16* __restrict__ v16, const _Float16* __restrict__ r16,
                 const float* __restrict__ relb, float* __restrict__ attn_out,
                 _Float16* __restrict__ hr16) {
  __shared__ __align__(16) _Float16 smA[128 * 128]; // Q, later attn (f16)
  __shared__ __align__(16) _Float16 smB[128 * 128]; // K, later V^T slab

  const int c = blockIdx.x, b = blockIdx.y;
  const int tid = threadIdx.x, wave = tid >> 5, lane = tid & 31;
  const int half = lane >> 4, ln = lane & 15;

  // stage Q and K (row-major, 128 x Z) via async DMA
  for (int i = tid; i < 128 * 16; i += 256) {
    int r = i >> 4, ch = i & 15;
    size_t grow = ((size_t)(c * 128 + r) * BATCH + b);
    async_ld16(lds_off_of(smA + r * 128 + ch * 8), q16 + grow * ZDIM + ch * 8);
    async_ld16(lds_off_of(smB + r * 128 + ch * 8), k16 + grow * ZDIM + ch * 8);
  }
  wait_async0();
  __syncthreads();

  // S = Q K^T  (q pre-scaled by Z^-0.5)
  v8f acc[8] = {};
#pragma unroll
  for (int kt = 0; kt < 4; kt++) {
    v16h a = load_frag(smA, wave * 16, 128, kt * 32);
#pragma unroll
    for (int nt = 0; nt < 8; nt++) {
      v16h bb = load_frag(smB, nt * 16, 128, kt * 32);
      acc[nt] = wmma_f16(a, bb, acc[nt]);
    }
  }

  // + rel-pos bias, row softmax (row = component j across 16 lanes of one half, 8 tiles)
#pragma unroll
  for (int j = 0; j < 8; j++) {
    int qrow = wave * 16 + half * 8 + j;
    float mx = -3.4e38f;
#pragma unroll
    for (int nt = 0; nt < 8; nt++) {
      int kcol = nt * 16 + ln;
      float s = acc[nt][j] + relb[1023 + kcol - qrow];
      acc[nt][j] = s;
      mx = fmaxf(mx, s);
    }
    for (int m = 1; m < 16; m <<= 1) mx = fmaxf(mx, __shfl_xor(mx, m, 32));
    float sum = 0.f;
#pragma unroll
    for (int nt = 0; nt < 8; nt++) {
      float e = __expf(acc[nt][j] - mx);
      acc[nt][j] = e;
      sum += e;
    }
    for (int m = 1; m < 16; m <<= 1) sum += __shfl_xor(sum, m, 32);
    float rinv = 1.f / sum;
#pragma unroll
    for (int nt = 0; nt < 8; nt++) acc[nt][j] *= rinv;
  }

  __syncthreads(); // all waves done reading Q/K tiles
  // emit attn (f32 output) and restage as f16 A-operand
#pragma unroll
  for (int j = 0; j < 8; j++) {
    int qrow = wave * 16 + half * 8 + j;
#pragma unroll
    for (int nt = 0; nt < 8; nt++) {
      int kcol = nt * 16 + ln;
      float a = acc[nt][j];
      attn_out[(((size_t)b * NCHUNK + c) * 128 + qrow) * 128 + kcol] = a;
      smA[qrow * 128 + kcol] = (_Float16)a;
    }
  }
  __syncthreads();

  v16h af[4];
#pragma unroll
  for (int kt = 0; kt < 4; kt++) af[kt] = load_frag(smA, wave * 16, 128, kt * 32);

  // h = attn @ V, H processed in 8 slabs of 128; V staged transposed (h-major) in smB
  for (int hb = 0; hb < 8; hb++) {
    __syncthreads(); // previous slab fully consumed
    for (int i = tid; i < 128 * 16; i += 256) {
      int k = i >> 4, ch = i & 15;
      size_t grow = ((size_t)(c * 128 + k) * BATCH + b);
      v8h vv = *(const v8h*)(v16 + grow * HDIM + hb * 128 + ch * 8);
#pragma unroll
      for (int e = 0; e < 8; e++) smB[(ch * 8 + e) * 128 + k] = vv[e];
    }
    __syncthreads();
#pragma unroll
    for (int nt = 0; nt < 8; nt++) {
      v8f hc = {};
#pragma unroll
      for (int kt = 0; kt < 4; kt++) {
        v16h bb = load_frag(smB, nt * 16, 128, kt * 32);
        hc = wmma_f16(af[kt], bb, hc);
      }
#pragma unroll
      for (int j = 0; j < 8; j++) {
        int qrow = wave * 16 + half * 8 + j;
        int hcol = hb * 128 + nt * 16 + ln;
        size_t grow = ((size_t)(c * 128 + qrow) * BATCH + b);
        float rv = (float)r16[grow * HDIM + hcol];
        hr16[grow * HDIM + hcol] = (_Float16)(hc[j] * rv);
      }
    }
  }
}

// ---------------------------------------------------------------- RMS norm finisher
__global__ __launch_bounds__(256)
void norm_kernel(const float* __restrict__ PN, const float* __restrict__ ns,
                 float* __restrict__ out) {
  int row = blockIdx.x * 8 + (threadIdx.x >> 5);
  int lane = threadIdx.x & 31;
  const float* p = PN + (size_t)row * DMODEL;
  float ss = 0.f;
  for (int i = lane; i < DMODEL; i += 32) {
    float v = p[i];
    ss += v * v;
  }
  for (int m = 1; m < 32; m <<= 1) ss += __shfl_xor(ss, m, 32);
  float scale = ns[0] * rsqrtf(ss * (1.f / DMODEL) + EPSV);
  float* o = out + (size_t)row * DMODEL;
  for (int i = lane; i < DMODEL; i += 32) o[i] = p[i] * scale;
}

// ---------------------------------------------------------------- launcher
extern "C" void kernel_launch(void* const* d_in, const int* in_sizes, int n_in,
                              void* d_out, int out_size, void* d_ws, size_t ws_size,
                              hipStream_t stream) {
  (void)in_sizes; (void)n_in; (void)out_size; (void)ws_size;
  const float* x      = (const float*)d_in[0];
  const float* edelta = (const float*)d_in[1];
  const float* ealpha = (const float*)d_in[2];
  const float* ebeta  = (const float*)d_in[3];
  const float* egamma = (const float*)d_in[4];
  const float* eomega = (const float*)d_in[5];
  const float* Wv     = (const float*)d_in[6];
  const float* bv     = (const float*)d_in[7];
  const float* Wmx    = (const float*)d_in[8];
  const float* bmx    = (const float*)d_in[9];
  const float* Wh     = (const float*)d_in[10];
  const float* bh     = (const float*)d_in[11];
  const float* qkg    = (const float*)d_in[12];
  const float* qkb    = (const float*)d_in[13];
  const float* relb   = (const float*)d_in[14];
  const float* nscal  = (const float*)d_in[15];

  float* out_main = (float*)d_out;                                   // (L,B,D)
  float* out_attn = (float*)d_out + (size_t)MROWS * DMODEL;          // (B,nc,128,128)

  char* wp = (char*)d_ws;
  auto alloc = [&](size_t bytes) {
    void* p = (void*)wp;
    wp += (bytes + 255) & ~(size_t)255;
    return p;
  };
  float*    fwdbuf = (float*)alloc((size_t)MROWS * DMODEL * 4);
  _Float16* x16    = (_Float16*)alloc((size_t)MROWS * DMODEL * 2);
  _Float16* mx16   = (_Float16*)alloc((size_t)MROWS * DMODEL * 2);
  _Float16* WvT    = (_Float16*)alloc((size_t)HDIM * DMODEL * 2);
  _Float16* WmxT   = (_Float16*)alloc((size_t)NTOTAL * DMODEL * 2);
  _Float16* WhT    = (_Float16*)alloc((size_t)DMODEL * HDIM * 2);
  _Float16* v16    = (_Float16*)alloc((size_t)MROWS * HDIM * 2);
  float*    Ubuf   = (float*)alloc((size_t)MROWS * DMODEL * 4);
  _Float16* q16    = (_Float16*)alloc((size_t)MROWS * ZDIM * 2);
  _Float16* k16    = (_Float16*)alloc((size_t)MROWS * ZDIM * 2);
  _Float16* r16    = (_Float16*)alloc((size_t)MROWS * HDIM * 2);
  float*    hxbuf  = (float*)alloc((size_t)MROWS * DMODEL * 4);
  _Float16* hr16   = (_Float16*)alloc((size_t)MROWS * HDIM * 2);
  float*    PN     = (float*)alloc((size_t)MROWS * DMODEL * 4);

  // precision conversion + weight transposes (N-major f16 for WMMA B operands)
  convert_to_f16<<<2048, 256, 0, stream>>>(x, x16, (size_t)MROWS * DMODEL);
  transpose_to_f16<<<1024, 256, 0, stream>>>(Wv, WvT, DMODEL, HDIM);
  transpose_to_f16<<<1024, 256, 0, stream>>>(Wmx, WmxT, DMODEL, NTOTAL);
  transpose_to_f16<<<1024, 256, 0, stream>>>(Wh, WhT, HDIM, DMODEL);

  // EMA scan -> mx16
  ema_kernel<<<(BATCH * DMODEL) / 128, 128, 0, stream>>>(x, edelta, ealpha, ebeta, egamma,
                                                         eomega, fwdbuf, mx16);

  // v = silu(x @ Wv + bv)
  gemm_kernel<0><<<dim3(HDIM / 64, MROWS / 64), 128, 0, stream>>>(
      x16, WvT, MROWS, HDIM, DMODEL, bv, v16,
      nullptr, nullptr, nullptr, nullptr, nullptr, nullptr, nullptr,
      nullptr, nullptr, nullptr, nullptr);

  // base = mx @ Wmx + bmx  -> u / q / k / r / hx (fused split + activations)
  gemm_kernel<1><<<dim3(NTOTAL / 64, MROWS / 64), 128, 0, stream>>>(
      mx16, WmxT, MROWS, NTOTAL, DMODEL, bmx, nullptr,
      Ubuf, q16, k16, r16, hxbuf, qkg, qkb,
      nullptr, nullptr, nullptr, nullptr);

  // chunked attention (+ attn output, fused *r into hr16)
  attn_kernel<<<dim3(NCHUNK, BATCH), 256, 0, stream>>>(q16, k16, v16, r16, relb,
                                                       out_attn, hr16);

  // pre-norm = x + u * (silu(hx + hr @ Wh + bh) - x)
  gemm_kernel<2><<<dim3(DMODEL / 64, MROWS / 64), 128, 0, stream>>>(
      hr16, WhT, MROWS, DMODEL, HDIM, bh, nullptr,
      nullptr, nullptr, nullptr, nullptr, nullptr, nullptr, nullptr,
      hxbuf, Ubuf, x, PN);

  // RMS norm
  norm_kernel<<<MROWS / 8, 256, 0, stream>>>(PN, nscal, out_main);
}
